// GGNN_83717502534014
// MI455X (gfx1250) — matrix operand order repaired and checked by reading, not compiled, not run on hardware
//
#include <hip/hip_runtime.h>
#include <hip/hip_bf16.h>
#include <math.h>

// ---------------- types ----------------
typedef __attribute__((ext_vector_type(16))) __bf16 v16bf;
typedef __attribute__((ext_vector_type(8)))  __bf16 bf16x8;
typedef __attribute__((ext_vector_type(8)))  float  v8f;
// exact param type of the async-to-LDS builtin on ROCm 7.2 (from diagnostics):
typedef int v4i_vec __attribute__((vector_size(4 * sizeof(int))));

// ---------------- problem dims ----------------
#define Hdim 512
#define NN   256
#define BB   32
#define EE   4
#define ADim 256
#define STEPS 5

// ---------------- GEMM tile ----------------
#define BM 128
#define BN 64
#define BK 32

// CDNA5 async global->LDS path (ASYNCcnt-tracked DMA), guarded so the file
// still compiles on toolchains without the builtin.
#if defined(__AMDGCN__) && __has_builtin(__builtin_amdgcn_global_load_async_to_lds_b128)
#define USE_ASYNC_LDS 1
#else
#define USE_ASYNC_LDS 0
#endif

#if USE_ASYNC_LDS
#define ASYNC_GPTR(p) ((__attribute__((address_space(1))) v4i_vec*)(p))
#define ASYNC_LPTR(p) ((__attribute__((address_space(3))) v4i_vec*)(p))
#endif

__device__ __forceinline__ void async_wait_all() {
#if USE_ASYNC_LDS
#if __has_builtin(__builtin_amdgcn_s_wait_asynccnt)
    __builtin_amdgcn_s_wait_asynccnt(0);
#else
    asm volatile("s_wait_asynccnt 0x0" ::: "memory");
#endif
#endif
}

// =====================================================================
// Elementwise helper kernels
// =====================================================================
__global__ void cvt_f32_bf16(const float* __restrict__ src, __bf16* __restrict__ dst, long n) {
    long i = (long)blockIdx.x * blockDim.x + threadIdx.x;
    if (i < n) dst[i] = (__bf16)src[i];
}

__global__ void embed_init(const int* __restrict__ ann_idx, const float* __restrict__ emb,
                           __bf16* __restrict__ ann_bf, float* __restrict__ prop_f,
                           __bf16* __restrict__ prop_bf) {
    int bn  = blockIdx.x;           // 0 .. B*N-1
    int idx = ann_idx[bn];
    for (int h = threadIdx.x; h < Hdim; h += blockDim.x) {
        float v = (h < ADim) ? emb[(long)idx * ADim + h] : 0.0f;
        prop_f [(long)bn * Hdim + h] = v;
        prop_bf[(long)bn * Hdim + h] = (__bf16)v;
        if (h < ADim) ann_bf[(long)bn * ADim + h] = (__bf16)v;
    }
}

__global__ void rp_mul(const float* __restrict__ r, const float* __restrict__ prop_f,
                       __bf16* __restrict__ rp, long n) {
    long i = (long)blockIdx.x * blockDim.x + threadIdx.x;
    if (i < n) rp[i] = (__bf16)(r[i] * prop_f[i]);
}

// =====================================================================
// Per-segment K-loop: pointer-increment addressing, async A-tile staging,
// transposed B staging, 2x2 WMMA 16x16x32 bf16 per wave.
// =====================================================================
__device__ __forceinline__ void gemm_k_loop(
    const __bf16* __restrict__ Aseg, int lda,      // already offset to (block rows, k=0)
    const __bf16* __restrict__ Bseg, int ldb,      // already offset to (k=segment base, block cols)
    int Klen,
    __bf16 (&As)[BM][40], __bf16 (&Bs)[BN][40],
    v8f (&acc)[2][2],
    int arow, int acol, int brow, int bcol, int lane, int wm, int wn)
{
    const __bf16* gA = Aseg + (long)arow * lda + acol;
    const __bf16* gB = Bseg + (long)brow * ldb + bcol;

    for (int kk = 0; kk < Klen; kk += BK) {
        // ---- stage A tile (BM x BK) row-major
#if USE_ASYNC_LDS
        __builtin_amdgcn_global_load_async_to_lds_b128(
            ASYNC_GPTR(gA),     ASYNC_LPTR(&As[arow][acol]),     0, 0);
        __builtin_amdgcn_global_load_async_to_lds_b128(
            ASYNC_GPTR(gA + 8), ASYNC_LPTR(&As[arow][acol + 8]), 0, 0);
#else
        {
            bf16x8 v0 = *(const bf16x8*)gA;
            bf16x8 v1 = *(const bf16x8*)(gA + 8);
            *(bf16x8*)&As[arow][acol]     = v0;
            *(bf16x8*)&As[arow][acol + 8] = v1;
        }
#endif
        // ---- stage B tile (BK x BN) transposed into Bs[col][k]
        {
            bf16x8 v = *(const bf16x8*)gB;
            if (kk + BK < Klen) __builtin_prefetch(gB + (long)BK * ldb, 0, 1);
#pragma unroll
            for (int j = 0; j < 8; ++j) Bs[bcol + j][brow] = v[j];
        }
        gA += BK;
        gB += (long)BK * ldb;

        async_wait_all();          // ASYNCcnt -> 0 before making LDS visible
        __syncthreads();

        // ---- fragments per ISA 16-bit operand layouts
        v16bf af[2], bfr[2];
        {
            const int r0 = wm * 32 + (lane & 15);
            const int kb = (lane >> 4) << 3;       // lanes 0-15: K{0..7,16..23}
#pragma unroll
            for (int mi = 0; mi < 2; ++mi) {
                const int r = r0 + mi * 16;
                bf16x8 lo = *(const bf16x8*)&As[r][kb];
                bf16x8 hi = *(const bf16x8*)&As[r][kb + 16];
#pragma unroll
                for (int i = 0; i < 8; ++i) { af[mi][i] = lo[i]; af[mi][8 + i] = hi[i]; }
            }
            const int c0  = wn * 32 + (lane & 15);
            const int kb2 = (lane >> 4) << 3;      // lanes 0-15: K{0..7,16..23}
#pragma unroll
            for (int ni = 0; ni < 2; ++ni) {
                const int c = c0 + ni * 16;
                bf16x8 lo = *(const bf16x8*)&Bs[c][kb2];
                bf16x8 hi = *(const bf16x8*)&Bs[c][kb2 + 16];
#pragma unroll
                for (int i = 0; i < 8; ++i) { bfr[ni][i] = lo[i]; bfr[ni][8 + i] = hi[i]; }
            }
        }
#pragma unroll
        for (int mi = 0; mi < 2; ++mi)
#pragma unroll
            for (int ni = 0; ni < 2; ++ni)
                acc[mi][ni] = __builtin_amdgcn_wmma_f32_16x16x32_bf16(
                    false, af[mi], false, bfr[ni], (short)0, acc[mi][ni], false, false);
        __syncthreads();
    }
}

// =====================================================================
// Generic WMMA bf16 GEMM with segmented-A (handles concatenated inputs)
//   C(z, m, n) = epilogue( sum_k A(m,k) * B(z, k, n) + bias(z, n) )
// modes: 0 = store bf16   1 = sigmoid -> f32
//        2 = tanh + GRU update (writes prop f32 + bf16)
//        3 = tanh -> f32
// cmap : 0 -> crow = m ; 1 -> crow = (m/256)*1024 + z*256 + m%256  (B,E,N layout)
// =====================================================================
__global__ __launch_bounds__(256)
void wmma_gemm(const __bf16* __restrict__ A0, const __bf16* __restrict__ A1,
               const __bf16* __restrict__ A2,
               int K0, int K1, int K2, int lda0, int lda1, int lda2, long aBS,
               const __bf16* __restrict__ Bmat, int ldb, long bBS,
               const float* __restrict__ bias, long biasBS,
               float* __restrict__ Cf, __bf16* __restrict__ Cbf, long cBS,
               int cmap, int mode,
               const float* __restrict__ zbuf, float* __restrict__ propf,
               __bf16* __restrict__ propbf)
{
    __shared__ __align__(16) __bf16 As[BM][40];   // 128 x 32 (+pad)
    __shared__ __align__(16) __bf16 Bs[BN][40];   // transposed: Bs[col][k]

    const int tid  = threadIdx.x;
    const int lane = tid & 31;
    const int wave = tid >> 5;
    const int wm   = wave & 3;          // 4 waves along M
    const int wn   = wave >> 2;         // 2 waves along N
    const int bm   = blockIdx.y, bn = blockIdx.x, z = blockIdx.z;

    const __bf16* Bz    = Bmat + (long)z * bBS + (long)bn * BN;
    const float*  biasZ = bias ? (bias + (long)z * biasBS) : nullptr;

    v8f acc[2][2] = {};

    const int arow = tid >> 1;          // 0..127
    const int acol = (tid & 1) << 4;    // 0 or 16
    const int brow = tid >> 3;          // 0..31
    const int bcol = (tid & 7) << 3;    // 0,8,..,56

    const long rowOff = (long)bm * BM;

    // Up to three K-segments (uniform control flow across the block).
    long kofs = 0;
    if (K0 > 0) {
        gemm_k_loop(A0 + (long)z * aBS + rowOff * lda0, lda0, Bz + kofs * ldb, ldb, K0,
                    As, Bs, acc, arow, acol, brow, bcol, lane, wm, wn);
        kofs += K0;
    }
    if (K1 > 0) {
        gemm_k_loop(A1 + (long)z * aBS + rowOff * lda1, lda1, Bz + kofs * ldb, ldb, K1,
                    As, Bs, acc, arow, acol, brow, bcol, lane, wm, wn);
        kofs += K1;
    }
    if (K2 > 0) {
        gemm_k_loop(A2 + (long)z * aBS + rowOff * lda2, lda2, Bz + kofs * ldb, ldb, K2,
                    As, Bs, acc, arow, acol, brow, bcol, lane, wm, wn);
    }

    // ---- epilogue (C layout: VGPR i -> M = i (lanes 0-15) / 8+i (lanes 16-31))
#pragma unroll
    for (int mi = 0; mi < 2; ++mi) {
#pragma unroll
        for (int ni = 0; ni < 2; ++ni) {
#pragma unroll
            for (int i = 0; i < 8; ++i) {
                const int mrow = (lane < 16) ? i : (8 + i);
                const int m = bm * BM + wm * 32 + mi * 16 + mrow;
                const int n = bn * BN + wn * 32 + ni * 16 + (lane & 15);
                float x = acc[mi][ni][i];
                if (biasZ) x += biasZ[n];
                long crow = (cmap == 0) ? (long)m
                          : ((long)(m >> 8) * (EE * NN) + (long)z * NN + (m & (NN - 1)));
                long cidx = (long)z * cBS + crow * Hdim + n;
                if (mode == 0) {
                    Cbf[cidx] = (__bf16)x;
                } else if (mode == 1) {
                    Cf[cidx] = 1.0f / (1.0f + __expf(-x));
                } else if (mode == 2) {
                    float h  = tanhf(x);
                    float zv = zbuf[cidx];
                    float pv = propf[cidx];
                    float np = fmaf(zv, h - pv, pv);      // (1-z)*p + z*h
                    propf[cidx]  = np;
                    propbf[cidx] = (__bf16)np;
                } else {
                    Cf[cidx] = tanhf(x);
                }
            }
        }
    }
}

// =====================================================================
// Host launch
// =====================================================================
extern "C" void kernel_launch(void* const* d_in, const int* in_sizes, int n_in,
                              void* d_out, int out_size, void* d_ws, size_t ws_size,
                              hipStream_t stream)
{
    const int*   annotation = (const int*)  d_in[0];
    const float* A     = (const float*)d_in[1];
    const float* emb   = (const float*)d_in[2];
    const float* W_in  = (const float*)d_in[3];
    const float* b_in  = (const float*)d_in[4];
    const float* W_out = (const float*)d_in[5];
    const float* b_out = (const float*)d_in[6];
    const float* Wr    = (const float*)d_in[7];
    const float* br    = (const float*)d_in[8];
    const float* Wz    = (const float*)d_in[9];
    const float* bz    = (const float*)d_in[10];
    const float* Wh    = (const float*)d_in[11];
    const float* bh    = (const float*)d_in[12];
    const float* Wo    = (const float*)d_in[13];
    const float* bo    = (const float*)d_in[14];
    float* out = (float*)d_out;

    char* ws = (char*)d_ws;
    size_t off = 0;
    auto carve = [&](size_t bytes) -> char* {
        char* p = ws + off;
        off = (off + bytes + 255) & ~(size_t)255;
        return p;
    };

    const long MR  = (long)BB * NN;            // 8192 rows
    const long KA  = (long)2 * EE * NN;        // 2048 adjacency cols

    __bf16* A_bf    = (__bf16*)carve(MR * KA * sizeof(__bf16));                 // 33.5 MB
    __bf16* Win_bf  = (__bf16*)carve((size_t)EE * Hdim * Hdim * sizeof(__bf16));
    __bf16* Wout_bf = (__bf16*)carve((size_t)EE * Hdim * Hdim * sizeof(__bf16));
    __bf16* Wrz_bf  = (__bf16*)carve((size_t)2 * 3 * Hdim * Hdim * sizeof(__bf16));
    __bf16* Wh_bf   = (__bf16*)carve((size_t)3 * Hdim * Hdim * sizeof(__bf16));
    __bf16* Wo_bf   = (__bf16*)carve((size_t)(Hdim + ADim) * Hdim * sizeof(__bf16));
    float*  brz     = (float*) carve((size_t)2 * Hdim * sizeof(float));
    __bf16* ann_bf  = (__bf16*)carve(MR * ADim * sizeof(__bf16));
    float*  prop_f  = (float*) carve(MR * Hdim * sizeof(float));
    __bf16* prop_bf = (__bf16*)carve(MR * Hdim * sizeof(__bf16));
    __bf16* ins_bf  = (__bf16*)carve((size_t)BB * EE * NN * Hdim * sizeof(__bf16));
    __bf16* outs_bf = (__bf16*)carve((size_t)BB * EE * NN * Hdim * sizeof(__bf16));
    __bf16* ain_bf  = (__bf16*)carve(MR * Hdim * sizeof(__bf16));
    __bf16* aout_bf = (__bf16*)carve(MR * Hdim * sizeof(__bf16));
    float*  rz_f    = (float*) carve((size_t)2 * MR * Hdim * sizeof(float));    // r then z
    __bf16* rp_bf   = (__bf16*)carve(MR * Hdim * sizeof(__bf16));

    auto cvt = [&](const float* s, __bf16* d, long n) {
        cvt_f32_bf16<<<dim3((unsigned)((n + 255) / 256)), 256, 0, stream>>>(s, d, n);
    };
    auto gemm = [&](dim3 grid,
                    const __bf16* a0, const __bf16* a1, const __bf16* a2,
                    int k0, int k1, int k2, int l0, int l1, int l2, long aBS,
                    const __bf16* Bm, int ldb, long bBS,
                    const float* bias, long biasBS,
                    float* cf, __bf16* cbf, long cBS, int cmap, int mode,
                    const float* zb, float* pf, __bf16* pbf) {
        wmma_gemm<<<grid, 256, 0, stream>>>(a0, a1, a2, k0, k1, k2, l0, l1, l2, aBS,
                                            Bm, ldb, bBS, bias, biasBS,
                                            cf, cbf, cBS, cmap, mode, zb, pf, pbf);
    };

    // ---- one-time conversions to bf16
    cvt(A,     A_bf,    MR * KA);
    cvt(W_in,  Win_bf,  (long)EE * Hdim * Hdim);
    cvt(W_out, Wout_bf, (long)EE * Hdim * Hdim);
    cvt(Wr,    Wrz_bf,                      (long)3 * Hdim * Hdim);
    cvt(Wz,    Wrz_bf + (long)3*Hdim*Hdim,  (long)3 * Hdim * Hdim);
    cvt(Wh,    Wh_bf,   (long)3 * Hdim * Hdim);
    cvt(Wo,    Wo_bf,   (long)(Hdim + ADim) * Hdim);
    (void)hipMemcpyAsync(brz,        br, Hdim * sizeof(float), hipMemcpyDeviceToDevice, stream);
    (void)hipMemcpyAsync(brz + Hdim, bz, Hdim * sizeof(float), hipMemcpyDeviceToDevice, stream);

    // ---- embedding gather + prop init
    embed_init<<<dim3((unsigned)MR), 256, 0, stream>>>(annotation, emb, ann_bf, prop_f, prop_bf);

    for (int step = 0; step < STEPS; ++step) {
        // ins[b,e,n,h] = prop @ W_in[e] + b_in[e]   (z = e, spread C layout)
        gemm(dim3(8, 64, 4), prop_bf, nullptr, nullptr, Hdim, 0, 0, Hdim, 0, 0, 0,
             Win_bf, Hdim, (long)Hdim * Hdim, b_in, Hdim,
             nullptr, ins_bf, 0, /*cmap*/1, /*mode*/0, nullptr, nullptr, nullptr);
        // outs
        gemm(dim3(8, 64, 4), prop_bf, nullptr, nullptr, Hdim, 0, 0, Hdim, 0, 0, 0,
             Wout_bf, Hdim, (long)Hdim * Hdim, b_out, Hdim,
             nullptr, outs_bf, 0, 1, 0, nullptr, nullptr, nullptr);
        // a_in[b] = A_in[b] @ ins[b]   (z = batch)
        gemm(dim3(8, 2, 32), A_bf, nullptr, nullptr, EE * NN, 0, 0, (int)KA, 0, 0, (long)NN * KA,
             ins_bf, Hdim, (long)EE * NN * Hdim, nullptr, 0,
             nullptr, ain_bf, (long)NN * Hdim, 0, 0, nullptr, nullptr, nullptr);
        // a_out[b] = A_out[b] @ outs[b]
        gemm(dim3(8, 2, 32), A_bf + EE * NN, nullptr, nullptr, EE * NN, 0, 0, (int)KA, 0, 0, (long)NN * KA,
             outs_bf, Hdim, (long)EE * NN * Hdim, nullptr, 0,
             nullptr, aout_bf, (long)NN * Hdim, 0, 0, nullptr, nullptr, nullptr);
        // r,z = sigmoid([a_in,a_out,prop] @ W{r,z} + b{r,z})   (z-axis = gate)
        gemm(dim3(8, 64, 2), ain_bf, aout_bf, prop_bf, Hdim, Hdim, Hdim, Hdim, Hdim, Hdim, 0,
             Wrz_bf, Hdim, (long)3 * Hdim * Hdim, brz, Hdim,
             rz_f, nullptr, MR * Hdim, 0, 1, nullptr, nullptr, nullptr);
        // rp = bf16(r * prop)
        rp_mul<<<dim3((unsigned)((MR * Hdim + 255) / 256)), 256, 0, stream>>>(
            rz_f, prop_f, rp_bf, MR * Hdim);
        // h_hat = tanh([a_in,a_out,rp] @ Wh + bh); prop = (1-z)*prop + z*h_hat (fused)
        gemm(dim3(8, 64, 1), ain_bf, aout_bf, rp_bf, Hdim, Hdim, Hdim, Hdim, Hdim, Hdim, 0,
             Wh_bf, Hdim, 0, bh, 0,
             nullptr, nullptr, 0, 0, 2, rz_f + MR * Hdim, prop_f, prop_bf);
    }

    // out = tanh([prop, ann] @ Wo + bo)
    gemm(dim3(8, 64, 1), prop_bf, ann_bf, nullptr, Hdim, ADim, 0, Hdim, ADim, 0, 0,
         Wo_bf, Hdim, 0, bo, 0,
         out, nullptr, 0, 0, 3, nullptr, nullptr, nullptr);
}
